// NSFP_627065225435
// MI455X (gfx1250) — matrix operand order repaired and checked by reading, not compiled, not run on hardware
//
#include <hip/hip_runtime.h>

typedef __attribute__((ext_vector_type(16))) _Float16 v16h;
typedef __attribute__((ext_vector_type(8)))  _Float16 v8h;
typedef __attribute__((ext_vector_type(8)))  float    v8f;

#define HSH     136      // LDS activation row stride in halves (272B, 16B-aligned, conflict-breaking)
#define TRUNC_D 2.0f
#define CT      256      // chamfer Y-tile
#define CSL     8        // chamfer Y-slices (occupancy)
#define WV      4        // waves per MLP block (64 rows/block)
#define LW      16384    // halves per hidden layer (128*128)

// ---- CDNA5 async global->LDS copy (ASYNCcnt path), inline asm for toolchain portability ----
__device__ __forceinline__ unsigned lds_off_u32(const void* p) {
    // flat LDS addresses map to LDS offset in the low 32 bits (ISA 10.2 aperture rules)
    return (unsigned)(unsigned long long)p;
}
__device__ __forceinline__ void async_copy_b128(unsigned ldsoff, const void* g) {
    asm volatile("global_load_async_to_lds_b128 %0, %1, off"
                 :: "v"(ldsoff), "v"((unsigned long long)g) : "memory");
}
__device__ __forceinline__ void wait_async0() {
    asm volatile("s_wait_asynccnt 0x0" ::: "memory");
}

// ---------------------------------------------------------------------------
// Repack Wh[l][128][128] f32 into WMMA B-fragment order, f16, lo/hi split:
// frag b = (l*8+t)*4+k occupies 512 halves:
//   [0:256)  = lo halves: lane*8 + h -> B[K = k*32 + kbase + h    ][t*16+col]
//   [256:512)= hi halves: lane*8 + h -> B[K = k*32 + kbase + 8 + h][t*16+col]
// with col = lane&15, kbase = (lane>>4)*16  (contiguous k order, same as the A fragments)
// ---------------------------------------------------------------------------
__global__ void nsfp_repack(const float* __restrict__ Wh, _Float16* __restrict__ w16, int layers)
{
    int b = blockIdx.x;            // ((l*8 + t)*4 + k)
    int k = b & 3;
    int t = (b >> 2) & 7;
    int l = b >> 5;
    if (l >= layers) return;
    int lane  = threadIdx.x;
    int col   = lane & 15;
    int kbase = (lane >> 4) * 16;
    _Float16* dst = w16 + (size_t)b * 512 + lane * 8;
    #pragma unroll
    for (int h = 0; h < 8; ++h) {
        int Klo = k * 32 + kbase + h;
        dst[h]       = (_Float16)Wh[((size_t)l * 128 + Klo)     * 128 + t * 16 + col];
        dst[256 + h] = (_Float16)Wh[((size_t)l * 128 + Klo + 8) * 128 + t * 16 + col];
    }
}

// ---------------------------------------------------------------------------
// Full MLP (3->128, 7x 128->128 WMMA, 128->3) + residual: out = in + sign*mlp(in)
// 4 waves / block, 16 rows each. Layer weights double-buffered in LDS via
// async copies overlapped with WMMAs. Two column tiles in flight per step
// (independent accumulator chains hide XDL latency); C starts from inline 0,
// bias folded into the ReLU writeback.
// ---------------------------------------------------------------------------
__global__ void __launch_bounds__(WV * 32)
nsfp_mlp(const float* __restrict__ in,                       // [N,3], residual base
         const float* __restrict__ W0, const float* __restrict__ b0,
         const _Float16* __restrict__ w16,                   // repacked hidden weights
         const float* __restrict__ bh,                       // [7][128]
         const float* __restrict__ Wl, const float* __restrict__ bl,
         float* __restrict__ out, int N, float sign)
{
    __shared__ alignas(16) _Float16 sW[2][LW];               // 2 x 32KB weight buffers
    __shared__ alignas(16) _Float16 Hbuf[WV][2][16 * HSH];   // per-wave activation ping-pong

    const int tid  = threadIdx.x;
    const int lane = tid & 31;
    const int wv   = tid >> 5;
    const int r0   = blockIdx.x * (WV * 16) + wv * 16;

    // ---- kick off async staging of layer-1 weights into sW[0] (32KB / 128 threads) ----
    {
        unsigned    d = lds_off_u32(&sW[0][0]) + tid * 16;
        const char* g = (const char*)w16 + tid * 16;
        #pragma unroll
        for (int i = 0; i < 16; ++i)
            async_copy_b128(d + i * (WV * 32 * 16), g + i * (WV * 32 * 16));
    }

    _Float16* Hcur = &Hbuf[wv][0][0];
    _Float16* Hnxt = &Hbuf[wv][1][0];

    // ---- layer 0: 3 -> 128 (VALU), ReLU; overlaps the async weight staging ----
    float w0r0[4], w0r1[4], w0r2[4], bb[4];
    #pragma unroll
    for (int cc = 0; cc < 4; ++cc) {
        int col = lane + 32 * cc;
        bb[cc]   = b0[col];
        w0r0[cc] = W0[col];
        w0r1[cc] = W0[128 + col];
        w0r2[cc] = W0[256 + col];
    }
    for (int r = 0; r < 16; ++r) {
        int g = r0 + r;
        float x0 = 0.f, x1 = 0.f, x2 = 0.f;
        if (g < N) { x0 = in[g*3+0]; x1 = in[g*3+1]; x2 = in[g*3+2]; }
        #pragma unroll
        for (int cc = 0; cc < 4; ++cc) {
            int col = lane + 32 * cc;
            float h = fmaf(x2, w0r2[cc], fmaf(x1, w0r1[cc], fmaf(x0, w0r0[cc], bb[cc])));
            Hcur[r * HSH + col] = (_Float16)fmaxf(h, 0.f);
        }
    }
    wait_async0();
    __syncthreads();

    // ---- layers 1..7: 128 -> 128, v_wmma_f32_16x16x32_f16, ReLU ----
    const int crow = lane & 15;
    const int khi  = lane >> 4;
    for (int l = 0; l < 7; ++l) {
        const _Float16* sWc = &sW[l & 1][0];

        // prefetch next layer's weights into the other buffer (ASYNCcnt overlap)
        if (l < 6) {
            unsigned    d = lds_off_u32(&sW[(l + 1) & 1][0]) + tid * 16;
            const char* g = (const char*)(w16 + (size_t)(l + 1) * LW) + tid * 16;
            #pragma unroll
            for (int i = 0; i < 16; ++i)
                async_copy_b128(d + i * (WV * 32 * 16), g + i * (WV * 32 * 16));
        }

        // hoist this wave's A fragments (16 rows x K=128) into registers once per layer
        v16h areg[4];
        #pragma unroll
        for (int k = 0; k < 4; ++k) {
            const _Float16* ap = Hcur + crow * HSH + k * 32 + khi * 16;
            v8h alo = *(const v8h*)(ap);
            v8h ahi = *(const v8h*)(ap + 8);
            #pragma unroll
            for (int h = 0; h < 8; ++h) { areg[k][h] = alo[h]; areg[k][8 + h] = ahi[h]; }
        }

        // preload the 8 per-tile biases for this layer (column = crow)
        float br[8];
        #pragma unroll
        for (int t = 0; t < 8; ++t) br[t] = bh[l * 128 + t * 16 + crow];

        // two column tiles in flight -> two independent WMMA chains
        for (int tp = 0; tp < 4; ++tp) {
            const int t0 = 2 * tp, t1 = 2 * tp + 1;
            v8f c0 = {};   // SRC2 = inline 0 on first WMMA of each chain
            v8f c1 = {};
            #pragma unroll
            for (int k = 0; k < 4; ++k) {
                const _Float16* fb0 = sWc + (t0 * 4 + k) * 512;
                const _Float16* fb1 = sWc + (t1 * 4 + k) * 512;
                v8h lo0 = *(const v8h*)(fb0 + lane * 8);
                v8h hi0 = *(const v8h*)(fb0 + 256 + lane * 8);
                v8h lo1 = *(const v8h*)(fb1 + lane * 8);
                v8h hi1 = *(const v8h*)(fb1 + 256 + lane * 8);
                v16h bf0, bf1;
                #pragma unroll
                for (int h = 0; h < 8; ++h) {
                    bf0[h] = lo0[h]; bf0[8 + h] = hi0[h];
                    bf1[h] = lo1[h]; bf1[8 + h] = hi1[h];
                }
                c0 = __builtin_amdgcn_wmma_f32_16x16x32_f16(false, areg[k], false, bf0,
                                                            (short)0, c0, false, false);
                c1 = __builtin_amdgcn_wmma_f32_16x16x32_f16(false, areg[k], false, bf1,
                                                            (short)0, c1, false, false);
            }
            #pragma unroll
            for (int v = 0; v < 8; ++v) {
                Hnxt[(khi * 8 + v) * HSH + t0 * 16 + crow] = (_Float16)fmaxf(c0[v] + br[t0], 0.f);
                Hnxt[(khi * 8 + v) * HSH + t1 * 16 + crow] = (_Float16)fmaxf(c1[v] + br[t1], 0.f);
            }
        }

        wait_async0();      // next layer's weights resident
        __syncthreads();    // all waves done reading sW[l&1] + own activations ordered
        _Float16* tmp = Hcur; Hcur = Hnxt; Hnxt = tmp;
    }

    // ---- final layer: 128 -> 3 (VALU) + residual ----
    if (lane < 16) {
        int g = r0 + lane;
        if (g < N) {
            float f0 = bl[0], f1 = bl[1], f2 = bl[2];
            const _Float16* hrow = Hcur + lane * HSH;
            for (int k2 = 0; k2 < 128; ++k2) {
                float hv = (float)hrow[k2];
                f0 = fmaf(hv, Wl[k2 * 3 + 0], f0);
                f1 = fmaf(hv, Wl[k2 * 3 + 1], f1);
                f2 = fmaf(hv, Wl[k2 * 3 + 2], f2);
            }
            out[g*3+0] = in[g*3+0] + sign * f0;
            out[g*3+1] = in[g*3+1] + sign * f1;
            out[g*3+2] = in[g*3+2] + sign * f2;
        }
    }
}

// ---------------------------------------------------------------------------
// Phase 1: all 4 chamfer directions fused (blockIdx.z) and Y split into CSL
// slices (blockIdx.y) for full-chip occupancy. 2 x-points/thread; Y tile in
// LDS (broadcast reads) + prefetch of the next tile. Writes per-(dir,slice,x)
// minimum squared distance (truncation happens after the slice-combine).
// ---------------------------------------------------------------------------
__global__ void __launch_bounds__(256)
nsfp_chamfer_min(const float* __restrict__ A, const float* __restrict__ B,
                 const float* __restrict__ C, const float* __restrict__ D,
                 int n, float* __restrict__ mind)            // [4][CSL][n]
{
    const float* X; const float* Y;
    switch (blockIdx.z) {
        case 0:  X = A; Y = B; break;    // pc0to1 -> pc1
        case 1:  X = B; Y = A; break;    // pc1    -> pc0to1
        case 2:  X = C; Y = D; break;    // est    -> pc0
        default: X = D; Y = C; break;    // pc0    -> est
    }
    const int s  = blockIdx.y;
    const int y0 = (int)(((long long)s       * n) / CSL);
    const int y1 = (int)(((long long)(s + 1) * n) / CSL);

    __shared__ float sy[CT * 3];
    int tid = threadIdx.x;
    int i0 = blockIdx.x * (2 * CT) + tid;
    int i1 = i0 + CT;
    bool va = (i0 < n), vb = (i1 < n);
    float a0 = 0.f, a1 = 0.f, a2 = 0.f, c0 = 0.f, c1 = 0.f, c2 = 0.f;
    if (va) { a0 = X[3*i0+0]; a1 = X[3*i0+1]; a2 = X[3*i0+2]; }
    if (vb) { c0 = X[3*i1+0]; c1 = X[3*i1+1]; c2 = X[3*i1+2]; }
    float dma = 3.402823e38f, dmc = 3.402823e38f;
    for (int base = y0; base < y1; base += CT) {
        int cnt = y1 - base; if (cnt > CT) cnt = CT;
        __syncthreads();
        if (tid < cnt) {
            sy[tid*3+0] = Y[(base + tid)*3+0];
            sy[tid*3+1] = Y[(base + tid)*3+1];
            sy[tid*3+2] = Y[(base + tid)*3+2];
        }
        if (base + CT + tid < y1)
            __builtin_prefetch(&Y[(base + CT + tid) * 3], 0, 0);   // global_prefetch_b8
        __syncthreads();
        for (int j = 0; j < cnt; ++j) {
            float v0 = sy[3*j+0], v1 = sy[3*j+1], v2 = sy[3*j+2];
            float d0 = a0 - v0, d1 = a1 - v1, d2 = a2 - v2;
            float e0 = c0 - v0, e1 = c1 - v1, e2 = c2 - v2;
            dma = fminf(dma, fmaf(d2, d2, fmaf(d1, d1, d0 * d0)));
            dmc = fminf(dmc, fmaf(e2, e2, fmaf(e1, e1, e0 * e0)));
        }
    }
    float* md = mind + ((size_t)blockIdx.z * CSL + s) * n;
    if (va) md[i0] = dma;
    if (vb) md[i1] = dmc;
}

// ---------------------------------------------------------------------------
// Phase 2: combine slice minima per (dir,x), apply truncation, deterministic
// block partial sums.
// ---------------------------------------------------------------------------
__global__ void __launch_bounds__(256)
nsfp_chamfer_reduce(const float* __restrict__ mind, int n, float* __restrict__ partial)
{
    __shared__ float wsum[256 / 32];
    int tid = threadIdx.x;
    int idx = blockIdx.x * 256 + tid;                  // over 4*n (dir-major)
    float v = 0.f;
    if (idx < 4 * n) {
        int dir = idx / n;
        int x   = idx - dir * n;
        const float* md = mind + (size_t)dir * CSL * n + x;
        float m = md[0];
        #pragma unroll
        for (int s = 1; s < CSL; ++s) m = fminf(m, md[(size_t)s * n]);
        v = (m < TRUNC_D) ? m : 0.f;
    }
    #pragma unroll
    for (int off = 16; off > 0; off >>= 1) v += __shfl_down(v, off, 32);
    if ((tid & 31) == 0) wsum[tid >> 5] = v;
    __syncthreads();
    if (tid == 0) {
        float s = 0.f;
        #pragma unroll
        for (int w = 0; w < 256 / 32; ++w) s += wsum[w];
        partial[blockIdx.x] = s;
    }
}

__global__ void nsfp_zero(float* __restrict__ p, int n)
{
    int i = blockIdx.x * blockDim.x + threadIdx.x;
    if (i < n) p[i] = 0.f;
}

__global__ void __launch_bounds__(32)
nsfp_finalize(const float* __restrict__ partial, int n, float* __restrict__ out, float scale)
{
    int lane = threadIdx.x;
    float s = 0.f;
    for (int i = lane; i < n; i += 32) s += partial[i];
    #pragma unroll
    for (int off = 16; off > 0; off >>= 1) s += __shfl_down(s, off, 32);
    if (lane == 0) out[0] = s * scale;
}

extern "C" void kernel_launch(void* const* d_in, const int* in_sizes, int n_in,
                              void* d_out, int out_size, void* d_ws, size_t ws_size,
                              hipStream_t stream)
{
    (void)n_in; (void)out_size; (void)ws_size;
    const float* pc0 = (const float*)d_in[0];
    const float* pc1 = (const float*)d_in[1];
    const float* W0  = (const float*)d_in[2];
    const float* b0  = (const float*)d_in[3];
    const float* Wh  = (const float*)d_in[4];
    const float* bh  = (const float*)d_in[5];
    const float* Wl  = (const float*)d_in[6];
    const float* bl  = (const float*)d_in[7];
    const float* W0i = (const float*)d_in[8];
    const float* b0i = (const float*)d_in[9];
    const float* Whi = (const float*)d_in[10];
    const float* bhi = (const float*)d_in[11];
    const float* Wli = (const float*)d_in[12];
    const float* bli = (const float*)d_in[13];

    const int N = in_sizes[0] / 3;
    const int layers = 7;
    const int NPART = 512;

    // workspace layout (all 16B-aligned offsets)
    float*    pc0to1   = (float*)d_ws;                       // N*3 f32
    float*    est      = pc0to1 + (size_t)N * 3;             // N*3 f32
    float*    partials = est + (size_t)N * 3;                // NPART f32
    float*    mind     = partials + NPART;                   // 4*CSL*N f32
    _Float16* w16f     = (_Float16*)(mind + (size_t)4 * CSL * N);
    _Float16* w16i     = w16f + (size_t)layers * 128 * 128;

    nsfp_repack<<<layers * 32, 32, 0, stream>>>(Wh,  w16f, layers);
    nsfp_repack<<<layers * 32, 32, 0, stream>>>(Whi, w16i, layers);

    int mgrid = (N + WV * 16 - 1) / (WV * 16);
    nsfp_mlp<<<mgrid, WV * 32, 0, stream>>>(pc0,    W0,  b0,  w16f, bh,  Wl,  bl,  pc0to1, N,  1.f);
    nsfp_mlp<<<mgrid, WV * 32, 0, stream>>>(pc0to1, W0i, b0i, w16i, bhi, Wli, bli, est,    N, -1.f);

    // fused 4-direction chamfer, Y sliced for occupancy: grid = (x-blocks, CSL, 4)
    dim3 cgrid((N + 2 * CT - 1) / (2 * CT), CSL, 4);
    nsfp_chamfer_min<<<cgrid, 256, 0, stream>>>(pc0to1, pc1, est, pc0, N, mind);

    nsfp_zero<<<(NPART + 255) / 256, 256, 0, stream>>>(partials, NPART);
    int rgrid = (4 * N + 255) / 256;                         // 313 <= NPART
    nsfp_chamfer_reduce<<<rgrid, 256, 0, stream>>>(mind, N, partials);

    nsfp_finalize<<<1, 32, 0, stream>>>(partials, NPART, (float*)d_out, 1.0f / (float)N);
}
